// Attention_35845797053156
// MI455X (gfx1250) — compile-verified, matrix operand order checked by reading, not run on hardware
//
#include <hip/hip_runtime.h>
#include <hip/hip_bf16.h>

typedef _Float16 v16h __attribute__((ext_vector_type(16)));
typedef _Float16 h8   __attribute__((ext_vector_type(8)));
typedef float    v8f  __attribute__((ext_vector_type(8)));

constexpr int kB  = 8;
constexpr int kH  = 128;
constexpr int kW  = 128;   // rows per slice (attention length)
constexpr int kC  = 512;   // channels
constexpr int kCK = 64;    // f/g projection dim

// LDS strides (in halfs). All chosen so rows are 16-byte aligned and the
// 16 rows touched by one fragment gather hit distinct bank groups.
constexpr int XS = 520;   // x_lds   [128][512] (+8 pad)   -> 133,120 B
constexpr int FS = 72;    // f/g_lds [128][64]  (+8 pad)   ->  18,432 B each
constexpr int BS = 136;   // beta    [128][128] (+8 pad)   ->  34,816 B
constexpr int HS = 136;   // hT      [2][16][128] (+8 pad) ->   8,704 B

union Frag16 { v16h v; h8 h[2]; };

// CDNA5 16-bit A-matrix 16x32 fragment (and mirrored B via row-major B^T):
// lanes 0-15 : row rtile+lane,    K = {kbase+0..7,  kbase+16..23}
// lanes 16-31: row rtile+lane-16, K = {kbase+8..15, kbase+24..31}
__device__ __forceinline__ v16h ld_frag(const _Float16* base, int stride,
                                        int rtile, int kbase, int lane) {
    const int r  = lane & 15;
    const int hi = (lane >> 4) & 1;
    const _Float16* p = base + (size_t)(rtile + r) * stride + kbase + hi * 8;
    Frag16 f;
    f.h[0] = *(const h8*)(p);        // 16B aligned
    f.h[1] = *(const h8*)(p + 16);   // 16B aligned
    return f.v;
}

__device__ __forceinline__ v8f wmma16(v16h a, v16h b, v8f c) {
    // D = A*B + C, fp16 inputs, fp32 accumulate
    return __builtin_amdgcn_wmma_f32_16x16x32_f16(false, a, false, b,
                                                  (short)0, c, false, false);
}

// One-time: convert weights fp32->f16 and transpose so columns (N) become
// rows (B-fragment friendly, contiguous along K).
__global__ void prep_weights(const float* __restrict__ wf,
                             const float* __restrict__ wg,
                             const float* __restrict__ wh,
                             _Float16* __restrict__ wfT,
                             _Float16* __restrict__ wgT,
                             _Float16* __restrict__ whT) {
    const int tid = blockIdx.x * blockDim.x + threadIdx.x;
    const int nt  = gridDim.x * blockDim.x;
    for (int i = tid; i < kC * kCK; i += nt) {
        const int k = i / kCK, n = i % kCK;        // wf[k][n]
        wfT[n * kC + k] = (_Float16)wf[i];
        wgT[n * kC + k] = (_Float16)wg[i];
    }
    for (int i = tid; i < kC * kC; i += nt) {
        const int k = i / kC, n = i % kC;          // wh[k][n]
        whT[n * kC + k] = (_Float16)wh[i];
    }
}

__global__ __launch_bounds__(256)
void attn_kernel(const float* __restrict__ x,
                 const _Float16* __restrict__ wfT,
                 const float* __restrict__ bf,
                 const _Float16* __restrict__ wgT,
                 const float* __restrict__ bg,
                 const _Float16* __restrict__ whT,
                 const float* __restrict__ bh,
                 const float* __restrict__ gammap,
                 float* __restrict__ out) {
    __shared__ _Float16 x_lds[kW * XS];
    __shared__ _Float16 f_lds[kW * FS];
    __shared__ _Float16 g_lds[kW * FS];
    __shared__ _Float16 beta_lds[kW * BS];
    __shared__ _Float16 hT_lds[2][16 * HS];   // double-buffered transpose tile

    const int tid   = threadIdx.x;
    const int wave  = tid >> 5;          // 8 waves
    const int lane  = tid & 31;
    const int r     = lane & 15;
    const int hi    = (lane >> 4) & 1;
    const int Mbase = wave * 16;         // rows owned by this wave

    const float gamma = gammap[0];
    const float* xrow = x   + (size_t)blockIdx.x * (kW * kC);
    float*       orow = out + (size_t)blockIdx.x * (kW * kC);

    // ---- Phase A: stage x slice (fp32 -> f16) into LDS -------------------
    for (int i4 = tid; i4 < (kW * kC) / 4; i4 += blockDim.x) {
        const int row = i4 >> 7;               // 128 float4 per row
        const int c4  = i4 & 127;
        const float4 vv = ((const float4*)xrow)[i4];
        _Float16* d = &x_lds[row * XS + c4 * 4];
        d[0] = (_Float16)vv.x; d[1] = (_Float16)vv.y;
        d[2] = (_Float16)vv.z; d[3] = (_Float16)vv.w;
    }
    __syncthreads();

    // ---- Hoist this wave's 16 x A-fragments into registers (128 VGPRs).
    // Reused by Phase B (f,g) and all 32 Phase-E h tiles.
    v16h xA[kC / 32];
    #pragma unroll
    for (int kt = 0; kt < kC / 32; ++kt)
        xA[kt] = ld_frag(x_lds, XS, Mbase, kt * 32, lane);

    // ---- Phase B: f = x@wf + bf ; g = x@wg + bg  (per-wave M tile) -------
    for (int nt = 0; nt < kCK / 16; ++nt) {
        v8f fa = {}; v8f ga = {};
        #pragma unroll
        for (int kt = 0; kt < kC / 32; ++kt) {
            const v16h wfv = ld_frag(wfT, kC, nt * 16, kt * 32, lane);
            const v16h wgv = ld_frag(wgT, kC, nt * 16, kt * 32, lane);
            fa = wmma16(xA[kt], wfv, fa);
            ga = wmma16(xA[kt], wgv, ga);
        }
        const float bfv = bf[nt * 16 + r];
        const float bgv = bg[nt * 16 + r];
        #pragma unroll
        for (int rr = 0; rr < 8; ++rr) {
            const int m = Mbase + rr + hi * 8;
            f_lds[m * FS + nt * 16 + r] = (_Float16)(fa[rr] + bfv);
            g_lds[m * FS + nt * 16 + r] = (_Float16)(ga[rr] + bgv);
        }
    }
    __syncthreads();

    // ---- Phase C: s = g @ f^T (K=64), row softmax, beta -> LDS -----------
    v8f sacc[8];
    for (int nt = 0; nt < 8; ++nt) {
        v8f acc = {};
        #pragma unroll
        for (int kt = 0; kt < 2; ++kt) {
            const v16h a = ld_frag(g_lds, FS, Mbase, kt * 32, lane);
            const v16h b = ld_frag(f_lds, FS, nt * 16, kt * 32, lane);
            acc = wmma16(a, b, acc);
        }
        sacc[nt] = acc;
    }
    #pragma unroll
    for (int rr = 0; rr < 8; ++rr) {
        // row Mbase+rr lives in lanes 0-15, row Mbase+rr+8 in lanes 16-31;
        // xor-shuffles with mask<16 reduce each half-wave (=one row) separately
        float m = sacc[0][rr];
        #pragma unroll
        for (int nt = 1; nt < 8; ++nt) m = fmaxf(m, sacc[nt][rr]);
        #pragma unroll
        for (int off = 1; off < 16; off <<= 1)
            m = fmaxf(m, __shfl_xor(m, off, 32));
        float e[8], sum = 0.f;
        #pragma unroll
        for (int nt = 0; nt < 8; ++nt) { e[nt] = __expf(sacc[nt][rr] - m); sum += e[nt]; }
        #pragma unroll
        for (int off = 1; off < 16; off <<= 1)
            sum += __shfl_xor(sum, off, 32);
        const float inv  = 1.0f / sum;
        const int   mrow = Mbase + rr + hi * 8;
        #pragma unroll
        for (int nt = 0; nt < 8; ++nt)
            beta_lds[mrow * BS + nt * 16 + r] = (_Float16)(e[nt] * inv);
    }
    __syncthreads();

    // ---- Phase D: keep beta A-fragments resident (K = 128 -> 4 tiles) ----
    v16h betaA[4];
    #pragma unroll
    for (int kt = 0; kt < 4; ++kt)
        betaA[kt] = ld_frag(beta_lds, BS, Mbase, kt * 32, lane);

    // ---- Phase E: stream h by 16-column tiles; fuse o = beta@h + epilogue.
    // Double-buffered hT scratch -> one barrier per tile. Safe because each
    // wave consumes its hT reads (s_wait_dscnt before the o-WMMA) before it
    // signals the next barrier, so buffer i is reusable two iterations later.
    for (int nt = 0; nt < kC / 16; ++nt) {
        _Float16* hT = hT_lds[nt & 1];
        // h tile: rows Mbase..Mbase+15, cols nt*16..nt*16+15, K over C=512.
        // A-fragments are register-resident: only global B loads remain here.
        v8f hacc = {};
        #pragma unroll
        for (int kt = 0; kt < kC / 32; ++kt) {
            const v16h b = ld_frag(whT, kC, nt * 16, kt * 32, lane);
            hacc = wmma16(xA[kt], b, hacc);
        }
        // transpose-store into hT[n_local][v] (B^T form for o GEMM)
        #pragma unroll
        for (int rr = 0; rr < 8; ++rr) {
            const int v = Mbase + rr + hi * 8;
            hT[r * HS + v] = (_Float16)hacc[rr];
        }
        __syncthreads();
        // o tile = beta @ h (K = W = 128)
        v8f oacc = {};
        #pragma unroll
        for (int kt = 0; kt < 4; ++kt) {
            const v16h b = ld_frag(hT, HS, 0, kt * 32, lane);
            oacc = wmma16(betaA[kt], b, oacc);
        }
        // epilogue: out = gamma*(o + bh) + x   (softmax rows sum to 1)
        const int   c   = nt * 16 + r;
        const float bhv = bh[c];
        #pragma unroll
        for (int rr = 0; rr < 8; ++rr) {
            const int m = Mbase + rr + hi * 8;
            orow[m * kC + c] = gamma * (oacc[rr] + bhv) + xrow[m * kC + c];
        }
    }
}

extern "C" void kernel_launch(void* const* d_in, const int* in_sizes, int n_in,
                              void* d_out, int out_size, void* d_ws, size_t ws_size,
                              hipStream_t stream) {
    const float* x     = (const float*)d_in[0];
    const float* wf    = (const float*)d_in[1];
    const float* bf    = (const float*)d_in[2];
    const float* wg    = (const float*)d_in[3];
    const float* bg    = (const float*)d_in[4];
    const float* wh    = (const float*)d_in[5];
    const float* bh    = (const float*)d_in[6];
    const float* gamma = (const float*)d_in[7];
    float* out = (float*)d_out;

    _Float16* wfT = (_Float16*)d_ws;           // 64*512 halfs
    _Float16* wgT = wfT + kC * kCK;            // 64*512 halfs
    _Float16* whT = wgT + kC * kCK;            // 512*512 halfs (total 640 KB)

    prep_weights<<<256, 256, 0, stream>>>(wf, wg, wh, wfT, wgT, whT);
    attn_kernel<<<kB * kH, 256, 0, stream>>>(x, wfT, bf, wgT, bg, whT, bh,
                                             gamma, out);
}
